// SpKBGATModified_e2t_37641093382709
// MI455X (gfx1250) — matrix operand order repaired and checked by reading, not compiled, not run on hardware
//
#include <hip/hip_runtime.h>
#include <math.h>

// ---------------------------------------------------------------------------
// SpKBGAT e2t forward on gfx1250 (MI455X), wave32 + v_wmma_f32_16x16x32_f16.
// 16 edges per wave = one WMMA M-tile; concat-GEMM done as 3 K-segments.
// Weights pre-packed into B-fragment lane layout; gathers staged in LDS (f16)
// with vectorized b128 loads / b64 LDS stores. Softmax-weighted scatter via
// global_atomic_add_f32.
// ---------------------------------------------------------------------------

#define K_ALPHA 0.2f
#define K_EPS   1e-12f

typedef _Float16 half_t;
typedef _Float16 v16h __attribute__((ext_vector_type(16)));
typedef _Float16 v8h  __attribute__((ext_vector_type(8)));
typedef _Float16 v4h  __attribute__((ext_vector_type(4)));
typedef float    v8f  __attribute__((ext_vector_type(8)));
typedef float    v4f  __attribute__((ext_vector_type(4)));

// ----- WMMA fragment loaders -----------------------------------------------
// A (16x32 f16) per ISA: lanes0-15 rows M=lane, K 0..7 & 16..23;
//                        lanes16-31 rows M=lane-16, K 8..15 & 24..31.
__device__ __forceinline__ v16h load_afrag(const half_t* A, int rowStride,
                                           int segOff, int kt, int lane)
{
    int m  = lane & 15;
    int kh = lane >> 4;
    const half_t* rp = A + m * rowStride + segOff + kt * 32;
    v8h lo = *(const v8h*)(rp + kh * 8);
    v8h hi = *(const v8h*)(rp + 16 + kh * 8);
    v16h a;
#pragma unroll
    for (int i = 0; i < 8; ++i) { a[i] = lo[i]; a[i + 8] = hi[i]; }
    return a;
}

// B blocks are pre-packed so lane's 16 halves are contiguous (2x b128 loads).
__device__ __forceinline__ v16h load_bfrag(const half_t* blk, int lane)
{
    const v8h* p = (const v8h*)blk + lane * 2;
    v8h lo = p[0];
    v8h hi = p[1];
    v16h b;
#pragma unroll
    for (int i = 0; i < 8; ++i) { b[i] = lo[i]; b[i + 8] = hi[i]; }
    return b;
}

// ----- 16xN GEMM pass over one packed weight matrix -------------------------
template<int SEGS, int KSEG, int NTILES>
__device__ __forceinline__ void gemm_pass(const half_t* A, const half_t* W,
                                          int lane, v8f* c)
{
    constexpr int KT = KSEG / 32;
#pragma unroll
    for (int nt = 0; nt < NTILES; ++nt)
#pragma unroll
        for (int r = 0; r < 8; ++r) c[nt][r] = 0.f;

#pragma unroll
    for (int seg = 0; seg < SEGS; ++seg) {
#pragma unroll
        for (int kt = 0; kt < KT; ++kt) {
            v16h a = load_afrag(A, SEGS * KSEG, seg * KSEG, kt, lane);
            const half_t* wb = W + (size_t)((seg * KT + kt) * NTILES) * 512;
#pragma unroll
            for (int nt = 0; nt < NTILES; ++nt) {
                v16h b = load_bfrag(wb + nt * 512, lane);
                c[nt] = __builtin_amdgcn_wmma_f32_16x16x32_f16(
                    false, a, false, b, (short)0, c[nt], false, false);
            }
        }
    }
}

// ----- attention logits -> e = exp(-leaky_relu(m1 @ att)) -------------------
// C layout: VGPR r, lanes0-15 -> row M=r, lanes16-31 -> row M=8+r; col = lane%16 + 16*nt.
template<int NTILES>
__device__ __forceinline__ void attn_e(const v8f* c, const float* __restrict__ att,
                                       int NV, int e0, int nE, int lane, float* ev)
{
    float pd[8];
#pragma unroll
    for (int r = 0; r < 8; ++r) pd[r] = 0.f;
    int nl = lane & 15;
#pragma unroll
    for (int nt = 0; nt < NTILES; ++nt) {
        int nn = nt * 16 + nl;
        float av = (nn < NV) ? att[nn] : 0.f;
#pragma unroll
        for (int r = 0; r < 8; ++r) pd[r] += c[nt][r] * av;
    }
#pragma unroll
    for (int off = 1; off < 16; off <<= 1) {
#pragma unroll
        for (int r = 0; r < 8; ++r) pd[r] += __shfl_xor(pd[r], off, 32);
    }
    int mb = (lane >> 4) * 8;
#pragma unroll
    for (int r = 0; r < 8; ++r) {
        float p  = pd[r];
        float lr = (p > 0.f) ? p : K_ALPHA * p;
        int   e  = e0 + mb + r;
        ev[r] = (e < nE) ? __expf(-lr) : 0.f;
    }
}

// ----- softmax-weighted scatter-add of one message matrix -------------------
template<int NTILES>
__device__ __forceinline__ void scatter_acc(const v8f* c, const float* ev,
                                            const int* __restrict__ idx,
                                            int e0, int nE,
                                            float* __restrict__ acc, int accStride, int colBase,
                                            float* __restrict__ rsarr, int rsStride, int rsCol,
                                            int NV, int lane)
{
    int nl = lane & 15;
    int mb = (lane >> 4) * 8;
    if (nl == 0) {
#pragma unroll
        for (int r = 0; r < 8; ++r) {
            int e = e0 + mb + r;
            if (e < nE)
                atomicAdd(rsarr + (size_t)idx[e] * rsStride + rsCol, ev[r]);
        }
    }
#pragma unroll
    for (int nt = 0; nt < NTILES; ++nt) {
        int nn = nt * 16 + nl;
        if (nn < NV) {
#pragma unroll
            for (int r = 0; r < 8; ++r) {
                int e = e0 + mb + r;
                if (e < nE)
                    atomicAdd(acc + (size_t)idx[e] * accStride + colBase + nn,
                              ev[r] * c[nt][r]);
            }
        }
    }
}

// ----- the edge-GAT kernel (one wave == 16 edges) ---------------------------
template<int SEGS, int KSEG, int KV, int NTILES, int NV, int WAVES>
__global__ __launch_bounds__(WAVES * 32)
void edge_gat_kernel(const int* __restrict__ srcI, const int* __restrict__ dstI,
                     const int* __restrict__ etI,
                     const float* __restrict__ T0, const float* __restrict__ T1,
                     const float* __restrict__ T2,
                     const half_t* __restrict__ W1p, const half_t* __restrict__ W2p,
                     const float* __restrict__ att,
                     float* __restrict__ acc1, float* __restrict__ rs1,
                     int accStride1, int colBase1, int rsStride1, int rsCol1,
                     float* __restrict__ acc2, float* __restrict__ rs2,
                     int accStride2, int colBase2, int rsStride2, int rsCol2,
                     int nEdges)
{
    __shared__ __align__(16) half_t Ast[WAVES][16][SEGS * KSEG];

    int wave = threadIdx.x >> 5;
    int lane = threadIdx.x & 31;
    int tile = blockIdx.x * WAVES + wave;
    int e0   = tile * 16;
    if (e0 >= nEdges) return;   // wave-uniform exit (EXEC stays all-ones)

    // prefetch the next tile's edge indices (global_prefetch_b8)
    if (lane == 0 && e0 + 16 < nEdges) {
        __builtin_prefetch(srcI + e0 + 16, 0, 1);
        __builtin_prefetch(dstI + e0 + 16, 0, 1);
        __builtin_prefetch(etI  + e0 + 16, 0, 1);
    }

    half_t* A = &Ast[wave][0][0];

    const int*   IDX[3] = { srcI, dstI, etI };
    const float* TBL[3] = { T0, T1, T2 };

    // Gather+convert edge feature rows (f32 -> f16) into LDS, zero-padded.
    // KV is a multiple of 4, so stage 4-wide: b128 global load + b64 LDS store.
#pragma unroll
    for (int seg = 0; seg < SEGS; ++seg) {
        for (int r = 0; r < 16; ++r) {
            int  e     = e0 + r;
            bool valid = (e < nEdges);
            int  row   = valid ? IDX[seg][e] : 0;
            const float* p = TBL[seg] + (size_t)row * KV;
            for (int k0 = lane * 4; k0 < KSEG; k0 += 128) {
                v4h h4;
                if (valid && k0 < KV) {
                    v4f f = *(const v4f*)(p + k0);
#pragma unroll
                    for (int i = 0; i < 4; ++i) h4[i] = (half_t)f[i];
                } else {
#pragma unroll
                    for (int i = 0; i < 4; ++i) h4[i] = (half_t)0.f;
                }
                *(v4h*)(A + r * (SEGS * KSEG) + seg * KSEG + k0) = h4;
            }
        }
    }

    v8f   c[NTILES];
    float ev[8];

    // pass 1: m1 = eh @ A1 -> attention -> scatter into entity accumulators
    gemm_pass<SEGS, KSEG, NTILES>(A, W1p, lane, c);
    attn_e<NTILES>(c, att, NV, e0, nEdges, lane, ev);
    scatter_acc<NTILES>(c, ev, srcI, e0, nEdges,
                        acc1, accStride1, colBase1, rs1, rsStride1, rsCol1, NV, lane);

    // pass 2: m2 = eh @ A2 -> scatter into type accumulators (same e)
    gemm_pass<SEGS, KSEG, NTILES>(A, W2p, lane, c);
    scatter_acc<NTILES>(c, ev, dstI, e0, nEdges,
                        acc2, accStride2, colBase2, rs2, rsStride2, rsCol2, NV, lane);
}

// ----- dense output: out = l2norm(X @ W + mask * elu(acc / (rs+eps))) -------
template<int WAVES>
__global__ __launch_bounds__(WAVES * 32)
void dense_out_kernel(const float* __restrict__ X, const half_t* __restrict__ Wp,
                      const float* __restrict__ acc, const float* __restrict__ rs,
                      const float* __restrict__ mask, float* __restrict__ out, int n)
{
    constexpr int KSEG = 128, KT = 4, NT = 13, NV = 200, KV = 100;
    __shared__ __align__(16) half_t Ast[WAVES][16][KSEG];

    int wave = threadIdx.x >> 5;
    int lane = threadIdx.x & 31;
    int tile = blockIdx.x * WAVES + wave;
    int r0   = tile * 16;
    if (r0 >= n) return;

    half_t* A = &Ast[wave][0][0];
    for (int r = 0; r < 16; ++r) {
        int  row   = r0 + r;
        bool valid = (row < n);
        const float* p = X + (size_t)row * KV;
        int k0 = lane * 4;
        v4h h4;
        if (valid && k0 < KV) {
            v4f f = *(const v4f*)(p + k0);
#pragma unroll
            for (int i = 0; i < 4; ++i) h4[i] = (half_t)f[i];
        } else {
#pragma unroll
            for (int i = 0; i < 4; ++i) h4[i] = (half_t)0.f;
        }
        *(v4h*)(A + r * KSEG + k0) = h4;
    }

    v8f c[NT];
#pragma unroll
    for (int nt = 0; nt < NT; ++nt)
#pragma unroll
        for (int r = 0; r < 8; ++r) c[nt][r] = 0.f;

#pragma unroll
    for (int kt = 0; kt < KT; ++kt) {
        v16h a = load_afrag(A, KSEG, 0, kt, lane);
        const half_t* wk = Wp + (size_t)(kt * NT) * 512;
#pragma unroll
        for (int nt = 0; nt < NT; ++nt) {
            v16h b = load_bfrag(wk + nt * 512, lane);
            c[nt] = __builtin_amdgcn_wmma_f32_16x16x32_f16(
                false, a, false, b, (short)0, c[nt], false, false);
        }
    }

    int nl = lane & 15;
    int mb = (lane >> 4) * 8;
    float mk[8], rv[8], ss[8];
#pragma unroll
    for (int r = 0; r < 8; ++r) {
        int row = r0 + mb + r;
        mk[r] = (row < n) ? mask[row] : 0.f;
        rv[r] = (row < n) ? rs[row]   : 0.f;
        ss[r] = 0.f;
    }
#pragma unroll
    for (int nt = 0; nt < NT; ++nt) {
        int nn = nt * 16 + nl;
#pragma unroll
        for (int r = 0; r < 8; ++r) {
            int row = r0 + mb + r;
            float v = c[nt][r];
            if (row < n && nn < NV) {
                float z = acc[(size_t)row * 208 + nn] / (rv[r] + K_EPS);
                z = (z > 0.f) ? z : (__expf(z) - 1.f);
                v += mk[r] * z;
            } else {
                v = 0.f;
            }
            c[nt][r] = v;
            ss[r] += v * v;
        }
    }
#pragma unroll
    for (int off = 1; off < 16; off <<= 1) {
#pragma unroll
        for (int r = 0; r < 8; ++r) ss[r] += __shfl_xor(ss[r], off, 32);
    }
    float inv[8];
#pragma unroll
    for (int r = 0; r < 8; ++r) inv[r] = 1.f / fmaxf(sqrtf(ss[r]), K_EPS);
#pragma unroll
    for (int nt = 0; nt < NT; ++nt) {
        int nn = nt * 16 + nl;
#pragma unroll
        for (int r = 0; r < 8; ++r) {
            int row = r0 + mb + r;
            if (row < n && nn < NV)
                out[(size_t)row * NV + nn] = c[nt][r] * inv[r];
        }
    }
}

// ----- helper kernels -------------------------------------------------------
__global__ void l2norm_kernel(const float* __restrict__ x, float* __restrict__ y,
                              int n, int d)
{
    int wv   = (blockIdx.x * blockDim.x + threadIdx.x) >> 5;
    int lane = threadIdx.x & 31;
    if (wv >= n) return;
    const float* p = x + (size_t)wv * d;
    float s = 0.f;
    for (int k0 = lane * 4; k0 < d; k0 += 128) {
        v4f f = *(const v4f*)(p + k0);
        s += f[0] * f[0] + f[1] * f[1] + f[2] * f[2] + f[3] * f[3];
    }
#pragma unroll
    for (int off = 1; off < 32; off <<= 1) s += __shfl_xor(s, off, 32);
    float inv = 1.f / fmaxf(sqrtf(s), K_EPS);
    float* q = y + (size_t)wv * d;
    for (int k0 = lane * 4; k0 < d; k0 += 128) {
        v4f f = *(const v4f*)(p + k0);
        v4f g;
#pragma unroll
        for (int i = 0; i < 4; ++i) g[i] = f[i] * inv;
        *(v4f*)(q + k0) = g;
    }
}

// Pack a (SEGS*KV, NV) f32 weight into WMMA B-fragment blocks of 512 f16.
__global__ void pack_kernel(const float* __restrict__ src, half_t* __restrict__ dst,
                            int SEGSv, int KT, int NT, int KV, int NV)
{
    int idx   = blockIdx.x * blockDim.x + threadIdx.x;
    int total = SEGSv * KT * NT * 512;
    if (idx >= total) return;
    int el  = idx & 511;
    int blk = idx >> 9;
    int nt  = blk % NT;  blk /= NT;
    int kt  = blk % KT;
    int seg = blk / KT;
    int lane = el >> 4;
    int i    = el & 15;
    int k = kt * 32 + (lane >> 4) * 8 + (i & 7) + (i >> 3) * 16;
    int n = nt * 16 + (lane & 15);
    float v = 0.f;
    if (k < KV && n < NV) v = src[(size_t)(seg * KV + k) * NV + n];
    dst[idx] = (half_t)v;
}

__global__ void relproj_kernel(const float* __restrict__ rel,
                               const float* __restrict__ W,
                               float* __restrict__ out, int nrel)
{
    int idx = blockIdx.x * blockDim.x + threadIdx.x;
    if (idx >= nrel * 200) return;
    int i = idx / 200, j = idx % 200;
    const float* r = rel + (size_t)i * 100;
    float s = 0.f;
    for (int k = 0; k < 100; ++k) s = fmaf(r[k], W[(size_t)k * 200 + j], s);
    out[idx] = s;
}

// layer-1 finalize: y[row, h*100+c] = elu(acc[row, h*112+c] / (rs[row,h]+eps))
__global__ void finalize1_kernel(const float* __restrict__ acc,
                                 const float* __restrict__ rs,
                                 float* __restrict__ y, int n)
{
    long idx = (long)blockIdx.x * blockDim.x + threadIdx.x;
    if (idx >= (long)n * 200) return;
    int row = (int)(idx / 200), cidx = (int)(idx % 200);
    int h = cidx / 100, cc = cidx % 100;
    float v = acc[(size_t)row * 224 + h * 112 + cc] / (rs[(size_t)row * 2 + h] + K_EPS);
    y[idx] = (v > 0.f) ? v : (__expf(v) - 1.f);
}

__global__ void mask_kernel(const int* __restrict__ batch,
                            float* __restrict__ m1, float* __restrict__ m2, int B)
{
    int i = blockIdx.x * blockDim.x + threadIdx.x;
    if (i >= B) return;
    m1[batch[i * 3 + 0]] = 1.f;
    m2[batch[i * 3 + 2]] = 1.f;
}

// ---------------------------------------------------------------------------
extern "C" void kernel_launch(void* const* d_in, const int* in_sizes, int n_in,
                              void* d_out, int out_size, void* d_ws, size_t ws_size,
                              hipStream_t stream)
{
    (void)n_in; (void)out_size; (void)ws_size;
    const int D = 100;

    const int*   batch = (const int*)d_in[0];
    const int*   elist = (const int*)d_in[1];
    const int*   etyp  = (const int*)d_in[2];
    const float* entE  = (const float*)d_in[3];
    const float* relE  = (const float*)d_in[4];
    const float* typE  = (const float*)d_in[5];
    const float* a1    = (const float*)d_in[6];
    const float* a2    = (const float*)d_in[7];
    const float* att   = (const float*)d_in[8];
    const float* Wrel  = (const float*)d_in[9];
    const float* a1o   = (const float*)d_in[10];
    const float* a2o   = (const float*)d_in[11];
    const float* atto  = (const float*)d_in[12];
    const float* Wen   = (const float*)d_in[13];
    const float* Wty   = (const float*)d_in[14];

    const int B    = in_sizes[0] / 3;
    const int E    = in_sizes[2];
    const int NENT = in_sizes[3] / D;
    const int NREL = in_sizes[4] / D;
    const int NTYP = in_sizes[5] / D;

    const int* srcI = elist;
    const int* dstI = elist + E;

    float* out  = (float*)d_out;
    float* o1   = out;
    float* o2   = out + (size_t)NENT * 200;
    float* orel = out + (size_t)(NENT + NTYP) * 200;

    // ---- workspace carve-up (bytes, 256-aligned) ----
    char*  wsb = (char*)d_ws;
    size_t off = 0;
    auto alloc = [&](size_t bytes) {
        size_t o = off;
        off = (off + bytes + 255) & ~(size_t)255;
        return o;
    };
    const size_t P1H = 3 * 4 * 7 * 512;       // halves per layer-1 pack
    const size_t P2H = 3 * 7 * 13 * 512;      // halves per layer-2 pack
    const size_t PDH = 1 * 4 * 13 * 512;      // halves per dense pack

    size_t oX1   = alloc((size_t)NENT * D * 4);
    size_t oX2   = alloc((size_t)NTYP * D * 4);
    size_t oW1P  = alloc(4 * P1H * 2);
    size_t oW2P  = alloc(2 * P2H * 2);
    size_t oWEP  = alloc(PDH * 2);
    size_t oWTP  = alloc(PDH * 2);
    size_t oACCE = alloc((size_t)NENT * 224 * 4);   // reused: L1 (stride 224) then L2 (stride 208)
    size_t oRSE  = alloc((size_t)NENT * 2 * 4);
    size_t oACCT = alloc((size_t)NTYP * 224 * 4);
    size_t oRST  = alloc((size_t)NTYP * 2 * 4);
    size_t oY1   = alloc((size_t)NENT * 200 * 4);
    size_t oY2   = alloc((size_t)NTYP * 200 * 4);
    size_t oM1   = alloc((size_t)NENT * 4);
    size_t oM2   = alloc((size_t)NTYP * 4);

    float*  X1   = (float*)(wsb + oX1);
    float*  X2   = (float*)(wsb + oX2);
    half_t* W1P  = (half_t*)(wsb + oW1P);
    half_t* W2P  = (half_t*)(wsb + oW2P);
    half_t* WEP  = (half_t*)(wsb + oWEP);
    half_t* WTP  = (half_t*)(wsb + oWTP);
    float*  ACCE = (float*)(wsb + oACCE);
    float*  RSE  = (float*)(wsb + oRSE);
    float*  ACCT = (float*)(wsb + oACCT);
    float*  RST  = (float*)(wsb + oRST);
    float*  Y1   = (float*)(wsb + oY1);
    float*  Y2   = (float*)(wsb + oY2);
    float*  M1   = (float*)(wsb + oM1);
    float*  M2   = (float*)(wsb + oM2);

    // ---- zero accumulators / masks ----
    hipMemsetAsync(ACCE, 0, (size_t)NENT * 224 * 4, stream);
    hipMemsetAsync(RSE,  0, (size_t)NENT * 2 * 4, stream);
    hipMemsetAsync(ACCT, 0, (size_t)NTYP * 224 * 4, stream);
    hipMemsetAsync(RST,  0, (size_t)NTYP * 2 * 4, stream);
    hipMemsetAsync(M1,   0, (size_t)NENT * 4, stream);
    hipMemsetAsync(M2,   0, (size_t)NTYP * 4, stream);

    // ---- normalize feature tables ----
    l2norm_kernel<<<(NENT + 7) / 8, 256, 0, stream>>>(entE, X1, NENT, D);
    l2norm_kernel<<<(NTYP + 7) / 8, 256, 0, stream>>>(typE, X2, NTYP, D);

    // ---- pack weights into WMMA B-fragment layout ----
    for (int h = 0; h < 2; ++h) {
        pack_kernel<<<(int)((P1H + 255) / 256), 256, 0, stream>>>(
            a1 + (size_t)h * 300 * 100, W1P + (size_t)(h * 2 + 0) * P1H, 3, 4, 7, 100, 100);
        pack_kernel<<<(int)((P1H + 255) / 256), 256, 0, stream>>>(
            a2 + (size_t)h * 300 * 100, W1P + (size_t)(h * 2 + 1) * P1H, 3, 4, 7, 100, 100);
    }
    pack_kernel<<<(int)((P2H + 255) / 256), 256, 0, stream>>>(a1o, W2P,        3, 7, 13, 200, 200);
    pack_kernel<<<(int)((P2H + 255) / 256), 256, 0, stream>>>(a2o, W2P + P2H,  3, 7, 13, 200, 200);
    pack_kernel<<<(int)((PDH + 255) / 256), 256, 0, stream>>>(Wen, WEP, 1, 4, 13, 100, 200);
    pack_kernel<<<(int)((PDH + 255) / 256), 256, 0, stream>>>(Wty, WTP, 1, 4, 13, 100, 200);

    // ---- out_rel = relation_emb @ W_rel (third output) ----
    relproj_kernel<<<(NREL * 200 + 255) / 256, 256, 0, stream>>>(relE, Wrel, orel, NREL);

    // ---- layer 1: two heads over edges ----
    const int ETILES = (E + 15) / 16;
    const int EBLK   = (ETILES + 3) / 4;
    for (int h = 0; h < 2; ++h) {
        edge_gat_kernel<3, 128, 100, 7, 100, 4><<<EBLK, 128, 0, stream>>>(
            srcI, dstI, etyp, X1, X2, relE,
            W1P + (size_t)(h * 2 + 0) * P1H, W1P + (size_t)(h * 2 + 1) * P1H,
            att + (size_t)h * 100,
            ACCE, RSE, 224, h * 112, 2, h,
            ACCT, RST, 224, h * 112, 2, h,
            E);
    }

    // ---- finalize layer-1 node features ----
    finalize1_kernel<<<(int)(((long)NENT * 200 + 255) / 256), 256, 0, stream>>>(ACCE, RSE, Y1, NENT);
    finalize1_kernel<<<(int)(((long)NTYP * 200 + 255) / 256), 256, 0, stream>>>(ACCT, RST, Y2, NTYP);

    // ---- re-zero accumulators for layer 2 ----
    hipMemsetAsync(ACCE, 0, (size_t)NENT * 224 * 4, stream);
    hipMemsetAsync(RSE,  0, (size_t)NENT * 2 * 4, stream);
    hipMemsetAsync(ACCT, 0, (size_t)NTYP * 224 * 4, stream);
    hipMemsetAsync(RST,  0, (size_t)NTYP * 2 * 4, stream);

    // ---- layer 2 (out_att) over edges ----
    edge_gat_kernel<3, 224, 200, 13, 200, 4><<<EBLK, 128, 0, stream>>>(
        srcI, dstI, etyp, Y1, Y2, orel,
        W2P, W2P + P2H, atto,
        ACCE, RSE, 208, 0, 1, 0,
        ACCT, RST, 208, 0, 1, 0,
        E);

    // ---- batch masks ----
    mask_kernel<<<(B + 255) / 256, 256, 0, stream>>>(batch, M1, M2, B);

    // ---- fused output projections + l2norm ----
    const int TTE = (NENT + 15) / 16;
    const int TTT = (NTYP + 15) / 16;
    dense_out_kernel<4><<<(TTE + 3) / 4, 128, 0, stream>>>(X1, WEP, ACCE, RSE, M1, o1, NENT);
    dense_out_kernel<4><<<(TTT + 3) / 4, 128, 0, stream>>>(X2, WTP, ACCT, RST, M2, o2, NTYP);
}
